// CaptchaCRNN_39298950758824
// MI455X (gfx1250) — matrix operand order repaired and checked by reading, not compiled, not run on hardware
//
#include <hip/hip_runtime.h>
#include <cstdint>
#include <cstddef>

// ---------------------------------------------------------------------------
// CDNA5 / gfx1250 CRNN implementation.
// All conv + GEMM work runs through v_wmma_f32_16x16x32_f16 (wave32, one wave
// per 16x64 output tile, 4x A-fragment reuse). f32 accumulate everywhere.
// ---------------------------------------------------------------------------

typedef _Float16 half_t;
typedef __attribute__((ext_vector_type(16))) _Float16 v16h;
typedef __attribute__((ext_vector_type(8)))  float    v8f;

// -------------------- weight convert: f32 [N][K] -> f16 [N][Kp] (zero pad) --
__global__ void cvt_w_kernel(const float* __restrict__ in, half_t* __restrict__ out,
                             int N, int K, int Kp) {
    int idx = blockIdx.x * blockDim.x + threadIdx.x;
    if (idx >= N * Kp) return;
    int n = idx / Kp, k = idx % Kp;
    out[idx] = (k < K) ? (half_t)in[(size_t)n * K + k] : (half_t)0.f;
}

__global__ void fill_zero_kernel(float* __restrict__ p, int n) {
    int i = blockIdx.x * blockDim.x + threadIdx.x;
    if (i < n) p[i] = 0.f;
}

// -------------------- implicit-GEMM conv via WMMA ---------------------------
// src: f32 NCHW, wgt: f16 [Co][Kp] (K = Ci*KH*KW, OIHW flat order),
// dst: f32 NCHW. Stride 1. One wave computes a 16(M) x 64(N) tile:
// 4 WMMAs per 32-wide K chunk sharing one A fragment.
template<int KH, int KW>
__global__ __launch_bounds__(32)
void conv_wmma_kernel(const float* __restrict__ src, const half_t* __restrict__ wgt,
                      const float* __restrict__ bias, float* __restrict__ dst,
                      int Ci, int IH, int IW, int Co,
                      int padH, int padW, int OH, int OW, int Kp) {
    const int lane = threadIdx.x;
    const int l15  = lane & 15;
    const int hs   = lane >> 4;          // 0: K 0-15, 1: K 16-31
    const int m0   = blockIdx.x * 16;
    const int n0   = blockIdx.y * 64;
    const int K    = Ci * KH * KW;

    // Decode this lane's A-matrix row (output pixel). M is a multiple of 16
    // for every layer in this net, so no M bounds check needed.
    int m  = m0 + l15;
    int ox = m % OW;  int t0 = m / OW;
    int oy = t0 % OH; int nb = t0 / OH;
    const float* sp = src + (size_t)nb * Ci * IH * IW;

    const half_t* br[4];
#pragma unroll
    for (int u = 0; u < 4; ++u)
        br[u] = wgt + (size_t)(n0 + u * 16 + l15) * Kp + hs * 16;

    const v8f vzero = {0.f,0.f,0.f,0.f,0.f,0.f,0.f,0.f};
    v8f acc[4];
#pragma unroll
    for (int u = 0; u < 4; ++u) acc[u] = vzero;

    for (int kc = 0; kc < Kp; kc += 32) {
        const int kbase = kc + hs * 16;
        v16h a;
#pragma unroll
        for (int j = 0; j < 8; ++j) {
            int k0 = kbase + j;        // element j
            int k1 = kbase + 8 + j;    // element 8 + j
            float v0 = 0.f, v1 = 0.f;
            if (k0 < K) {
                int kx = k0 % KW; int r = k0 / KW; int ky = r % KH; int ci = r / KH;
                int iy = oy - padH + ky, ix = ox - padW + kx;
                if ((unsigned)iy < (unsigned)IH && (unsigned)ix < (unsigned)IW)
                    v0 = sp[((size_t)ci * IH + iy) * IW + ix];
            }
            if (k1 < K) {
                int kx = k1 % KW; int r = k1 / KW; int ky = r % KH; int ci = r / KH;
                int iy = oy - padH + ky, ix = ox - padW + kx;
                if ((unsigned)iy < (unsigned)IH && (unsigned)ix < (unsigned)IW)
                    v1 = sp[((size_t)ci * IH + iy) * IW + ix];
            }
            a[j]     = (half_t)v0;
            a[j + 8] = (half_t)v1;
        }
#pragma unroll
        for (int u = 0; u < 4; ++u) {
            __builtin_prefetch((const void*)(br[u] + kc + 32), 0, 1);
            v16h b = *(const v16h*)(br[u] + kc);
            acc[u] = __builtin_amdgcn_wmma_f32_16x16x32_f16(
                false, a, false, b, (short)0, acc[u], false, false);
        }
    }

#pragma unroll
    for (int u = 0; u < 4; ++u) {
        int n = n0 + u * 16 + l15;
        float bv = bias[n];
#pragma unroll
        for (int r = 0; r < 8; ++r) {
            int mm  = m0 + r + hs * 8;
            int ox2 = mm % OW;  int t2  = mm / OW;
            int oy2 = t2 % OH;  int nb2 = t2 / OH;
            dst[(((size_t)nb2 * Co + n) * OH + oy2) * OW + ox2] = acc[u][r] + bv;
        }
    }
}

// -------------------- dense GEMM via WMMA -----------------------------------
// D[M][N] = A[M][K](f32) * Bw[N][K](f16, row = contiguous K) + Cin + bias1+bias2
// M % 16 == 0, N % 64 == 0, K % 32 == 0.
__global__ __launch_bounds__(32)
void gemm_wmma_kernel(const float* __restrict__ A, const half_t* __restrict__ Bw,
                      const float* __restrict__ Cin,
                      const float* __restrict__ bias1, const float* __restrict__ bias2,
                      float* __restrict__ D, int N, int K) {
    const int lane = threadIdx.x;
    const int l15  = lane & 15;
    const int hs   = lane >> 4;
    const int m0   = blockIdx.x * 16;
    const int n0   = blockIdx.y * 64;

    const float*  arow = A + (size_t)(m0 + l15) * K;
    const half_t* br[4];
#pragma unroll
    for (int u = 0; u < 4; ++u)
        br[u] = Bw + (size_t)(n0 + u * 16 + l15) * K + hs * 16;

    const v8f vzero = {0.f,0.f,0.f,0.f,0.f,0.f,0.f,0.f};
    v8f acc[4];
#pragma unroll
    for (int u = 0; u < 4; ++u) acc[u] = vzero;
    if (Cin) {
#pragma unroll
        for (int u = 0; u < 4; ++u)
#pragma unroll
            for (int r = 0; r < 8; ++r)
                acc[u][r] = Cin[(size_t)(m0 + r + hs * 8) * N + (n0 + u * 16 + l15)];
    }

    for (int kc = 0; kc < K; kc += 32) {
        const int kbase = kc + hs * 16;
        v16h a;
#pragma unroll
        for (int j = 0; j < 8; ++j) {
            a[j]     = (half_t)arow[kbase + j];
            a[j + 8] = (half_t)arow[kbase + 8 + j];
        }
#pragma unroll
        for (int u = 0; u < 4; ++u) {
            v16h b = *(const v16h*)(br[u] + kc);
            acc[u] = __builtin_amdgcn_wmma_f32_16x16x32_f16(
                false, a, false, b, (short)0, acc[u], false, false);
        }
    }

#pragma unroll
    for (int u = 0; u < 4; ++u) {
        int n = n0 + u * 16 + l15;
        float bv = 0.f;
        if (bias1) bv += bias1[n];
        if (bias2) bv += bias2[n];
#pragma unroll
        for (int r = 0; r < 8; ++r)
            D[(size_t)(m0 + r + hs * 8) * N + n] = acc[u][r] + bv;
    }
}

// -------------------- maxpool (VALID) ---------------------------------------
__global__ void maxpool_kernel(const float* __restrict__ src, float* __restrict__ dst,
                               int C, int IH, int IW, int wH, int wW, int sH, int sW,
                               int OH, int OW, int total) {
    int idx = blockIdx.x * blockDim.x + threadIdx.x;
    if (idx >= total) return;
    int ox = idx % OW; int t = idx / OW;
    int oy = t % OH;   t /= OH;
    int c  = t % C;    int nb = t / C;
    const float* p = src + ((size_t)nb * C + c) * IH * IW;
    float m = -__builtin_inff();
    for (int dy = 0; dy < wH; ++dy)
        for (int dx = 0; dx < wW; ++dx)
            m = fmaxf(m, p[(oy * sH + dy) * IW + (ox * sW + dx)]);
    dst[idx] = m;
}

// -------------------- batchnorm: stats then apply ---------------------------
__global__ void bn_stats_kernel(const float* __restrict__ x, float* __restrict__ scale,
                                float* __restrict__ shift, const float* __restrict__ gamma,
                                const float* __restrict__ beta, int NB, int C, int HW) {
    __shared__ float s1[256], s2[256];
    int c = blockIdx.x, tid = threadIdx.x;
    int total = NB * HW;
    float sum = 0.f, sq = 0.f;
    for (int i = tid; i < total; i += 256) {
        int n = i / HW, p = i % HW;
        float v = x[((size_t)n * C + c) * HW + p];
        sum += v; sq += v * v;
    }
    s1[tid] = sum; s2[tid] = sq;
    __syncthreads();
    for (int s = 128; s > 0; s >>= 1) {
        if (tid < s) { s1[tid] += s1[tid + s]; s2[tid] += s2[tid + s]; }
        __syncthreads();
    }
    if (tid == 0) {
        float mean = s1[0] / total;
        float var  = s2[0] / total - mean * mean;
        float sc   = gamma[c] * rsqrtf(var + 1e-5f);
        scale[c] = sc;
        shift[c] = beta[c] - mean * sc;
    }
}

__global__ void bn_apply_kernel(float* __restrict__ x, const float* __restrict__ scale,
                                const float* __restrict__ shift, int C, int HW, int total) {
    int idx = blockIdx.x * blockDim.x + threadIdx.x;
    if (idx >= total) return;
    int c = (idx / HW) % C;
    x[idx] = x[idx] * scale[c] + shift[c];
}

// -------------------- conv7 (B,512,3,15) -> seq [T=15][B=64][D=1024] --------
// seq[t][b][2c+h] = x7[b][c][h][t], h in {0,1}
__global__ void build_seq_kernel(const float* __restrict__ x7, float* __restrict__ seq) {
    int idx = blockIdx.x * blockDim.x + threadIdx.x;
    const int total = 15 * 64 * 1024;
    if (idx >= total) return;
    int d = idx & 1023; int t2 = idx >> 10;
    int b = t2 % 64;    int t  = t2 / 64;
    int c = d >> 1, h = d & 1;
    seq[idx] = x7[(((size_t)b * 512 + c) * 3 + h) * 15 + t];
}

// -------------------- fused LSTM gate update (both directions) --------------
// xg rows laid out [t][b][4H]; blockIdx.y = direction (0=f, 1=b).
__global__ void lstm_gates_kernel(const float* __restrict__ xgF, const float* __restrict__ xgB,
                                  float* __restrict__ cF, float* __restrict__ cB,
                                  float* __restrict__ hF, float* __restrict__ hB,
                                  float* __restrict__ out, int tF, int tB, int T, int H, int B) {
    int idx = blockIdx.x * blockDim.x + threadIdx.x;
    if (idx >= B * H) return;
    int b = idx / H, j = idx % H;
    int dir = blockIdx.y;
    int t = dir ? tB : tF;
    const float* g = (dir ? xgB : xgF) + ((size_t)t * B + b) * (4 * H);
    float* c = (dir ? cB : cF) + idx;
    float* h = (dir ? hB : hF) + idx;
    float i_ = 1.f / (1.f + expf(-g[j]));
    float f_ = 1.f / (1.f + expf(-g[H + j]));
    float g_ = tanhf(g[2 * H + j]);
    float o_ = 1.f / (1.f + expf(-g[3 * H + j]));
    float cc = f_ * (*c) + i_ * g_;
    float hh = o_ * tanhf(cc);
    *c = cc; *h = hh;
    out[((size_t)b * T + t) * (2 * H) + dir * H + j] = hh;
}

// ===========================================================================
extern "C" void kernel_launch(void* const* d_in, const int* in_sizes, int n_in,
                              void* d_out, int out_size, void* d_ws, size_t ws_size,
                              hipStream_t stream) {
    (void)in_sizes; (void)n_in; (void)out_size; (void)ws_size;

    const float* x = (const float*)d_in[0];
    const float *w[8], *bb[8];
    for (int i = 1; i <= 7; ++i) { w[i] = (const float*)d_in[2*i-1]; bb[i] = (const float*)d_in[2*i]; }
    const float* gamma = (const float*)d_in[15];
    const float* beta  = (const float*)d_in[16];
    const float* Wih_f = (const float*)d_in[17]; const float* Whh_f = (const float*)d_in[18];
    const float* bih_f = (const float*)d_in[19]; const float* bhh_f = (const float*)d_in[20];
    const float* Wih_b = (const float*)d_in[21]; const float* Whh_b = (const float*)d_in[22];
    const float* bih_b = (const float*)d_in[23]; const float* bhh_b = (const float*)d_in[24];
    float* out = (float*)d_out;

    // ---- workspace bump allocator (deterministic fixed layout) ----
    uintptr_t cur = (uintptr_t)d_ws;
    auto alloc = [&](size_t bytes) -> void* {
        uintptr_t p = (cur + 255) & ~(uintptr_t)255;
        cur = p + bytes;
        return (void*)p;
    };

    float* big0 = (float*)alloc((size_t)64 * 64 * 64 * 256 * 4);   // 256 MiB ping
    float* big1 = (float*)alloc((size_t)64 * 128 * 32 * 128 * 4);  // 128 MiB pong

    // conv specs: Co, Ci, KH, KW
    const int spec[7][4] = {{64,1,3,3},{128,64,3,3},{256,128,3,3},{256,256,3,3},
                            {512,256,3,3},{512,512,3,3},{512,512,2,2}};
    half_t* wf[8]; int Kp[8];
    for (int i = 0; i < 7; ++i) {
        int Co = spec[i][0], Ci = spec[i][1], KH = spec[i][2], KW = spec[i][3];
        int K = Ci * KH * KW;
        Kp[i+1] = (K + 31) & ~31;
        wf[i+1] = (half_t*)alloc((size_t)Co * Kp[i+1] * 2);
        int tot = Co * Kp[i+1];
        cvt_w_kernel<<<(tot + 255) / 256, 256, 0, stream>>>(w[i+1], wf[i+1], Co, K, Kp[i+1]);
    }
    half_t* wih16f = (half_t*)alloc((size_t)1024 * 1024 * 2);
    half_t* wih16b = (half_t*)alloc((size_t)1024 * 1024 * 2);
    half_t* whh16f = (half_t*)alloc((size_t)1024 * 256 * 2);
    half_t* whh16b = (half_t*)alloc((size_t)1024 * 256 * 2);
    cvt_w_kernel<<<(1024*1024 + 255)/256, 256, 0, stream>>>(Wih_f, wih16f, 1024, 1024, 1024);
    cvt_w_kernel<<<(1024*1024 + 255)/256, 256, 0, stream>>>(Wih_b, wih16b, 1024, 1024, 1024);
    cvt_w_kernel<<<(1024*256  + 255)/256, 256, 0, stream>>>(Whh_f, whh16f, 1024, 256, 256);
    cvt_w_kernel<<<(1024*256  + 255)/256, 256, 0, stream>>>(Whh_b, whh16b, 1024, 256, 256);

    float* seq   = (float*)alloc((size_t)15 * 64 * 1024 * 4);
    float* xgF   = (float*)alloc((size_t)15 * 64 * 1024 * 4);
    float* xgB   = (float*)alloc((size_t)15 * 64 * 1024 * 4);
    float* bnsc  = (float*)alloc(512 * 4);
    float* bnsh  = (float*)alloc(512 * 4);
    float* state = (float*)alloc((size_t)4 * 64 * 256 * 4);
    float* hF = state, *hB = state + 16384, *cF = state + 32768, *cB = state + 49152;

    auto conv = [&](int i, const float* src, float* dst, int IH, int IW, int padded) {
        int Co = spec[i-1][0], Ci = spec[i-1][1], KH = spec[i-1][2], KW = spec[i-1][3];
        int OH = padded ? IH : IH - KH + 1;
        int OW = padded ? IW : IW - KW + 1;
        int pad = padded ? 1 : 0;
        dim3 g((unsigned)(64 * OH * OW / 16), (unsigned)(Co / 64));
        if (KH == 3)
            conv_wmma_kernel<3,3><<<g, 32, 0, stream>>>(src, wf[i], bb[i], dst,
                Ci, IH, IW, Co, pad, pad, OH, OW, Kp[i]);
        else
            conv_wmma_kernel<2,2><<<g, 32, 0, stream>>>(src, wf[i], bb[i], dst,
                Ci, IH, IW, Co, pad, pad, OH, OW, Kp[i]);
    };
    auto pool = [&](const float* s, float* d, int C, int IH, int IW,
                    int wH, int wW, int sH, int sW) {
        int OH = (IH - wH) / sH + 1, OW = (IW - wW) / sW + 1;
        int tot = 64 * C * OH * OW;
        maxpool_kernel<<<(tot + 255)/256, 256, 0, stream>>>(s, d, C, IH, IW, wH, wW, sH, sW, OH, OW, tot);
    };
    auto bn = [&](float* buf) {  // (64,512,8,32), in place
        bn_stats_kernel<<<512, 256, 0, stream>>>(buf, bnsc, bnsh, gamma, beta, 64, 512, 8*32);
        int tot = 64 * 512 * 8 * 32;
        bn_apply_kernel<<<(tot + 255)/256, 256, 0, stream>>>(buf, bnsc, bnsh, 512, 8*32, tot);
    };

    // ---- CNN chain ----
    conv(1, x,    big0, 64, 256, 1);                 // (64,64,64,256)
    pool(big0, big1, 64, 64, 256, 2, 2, 2, 2);       // (64,64,32,128)
    conv(2, big1, big0, 32, 128, 1);                 // (64,128,32,128)
    pool(big0, big1, 128, 32, 128, 2, 2, 2, 2);      // (64,128,16,64)
    conv(3, big1, big0, 16, 64, 1);                  // (64,256,16,64)
    conv(4, big0, big1, 16, 64, 1);                  // (64,256,16,64)
    pool(big1, big0, 256, 16, 64, 1, 2, 2, 2);       // (64,256,8,32)
    conv(5, big0, big1, 8, 32, 1);                   // (64,512,8,32)
    bn(big1);
    conv(6, big1, big0, 8, 32, 1);                   // (64,512,8,32)
    bn(big0);
    pool(big0, big1, 512, 8, 32, 1, 2, 2, 2);        // (64,512,4,16)
    conv(7, big1, big0, 4, 16, 0);                   // (64,512,3,15) VALID

    // ---- sequence + input-gate GEMMs (M=960, N=1024, K=1024) ----
    {
        int tot = 15 * 64 * 1024;
        build_seq_kernel<<<(tot + 255)/256, 256, 0, stream>>>(big0, seq);
    }
    gemm_wmma_kernel<<<dim3(60, 16), 32, 0, stream>>>(seq, wih16f, nullptr, bih_f, bhh_f, xgF, 1024, 1024);
    gemm_wmma_kernel<<<dim3(60, 16), 32, 0, stream>>>(seq, wih16b, nullptr, bih_b, bhh_b, xgB, 1024, 1024);

    // ---- BiLSTM recurrence: 15 steps, each a 64x1024x256 WMMA GEMM ----
    fill_zero_kernel<<<(4 * 64 * 256 + 255)/256, 256, 0, stream>>>(state, 4 * 64 * 256);
    for (int s = 0; s < 15; ++s) {
        int tF = s, tB = 14 - s;
        float* gF = xgF + (size_t)tF * 64 * 1024;
        float* gB = xgB + (size_t)tB * 64 * 1024;
        gemm_wmma_kernel<<<dim3(4, 16), 32, 0, stream>>>(hF, whh16f, gF, nullptr, nullptr, gF, 1024, 256);
        gemm_wmma_kernel<<<dim3(4, 16), 32, 0, stream>>>(hB, whh16b, gB, nullptr, nullptr, gB, 1024, 256);
        lstm_gates_kernel<<<dim3(64, 2), 256, 0, stream>>>(xgF, xgB, cF, cB, hF, hB, out, tF, tB, 15, 256, 64);
    }
}
